// MambaSSM_27582279975001
// MI455X (gfx1250) — compile-verified, matrix operand order checked by reading, not run on hardware
//
#include <hip/hip_runtime.h>
#include <hip/hip_bf16.h>
#include <math.h>

typedef __attribute__((ext_vector_type(16))) _Float16 v16h;
typedef __attribute__((ext_vector_type(8)))  float    v8f;
typedef __attribute__((ext_vector_type(2)))  _Float16 h2v;

#define DMODEL 1024
#define DINNER 2048
#define NSTATE 16
#define NBATCH 4
#define LSEQ   2048
#define MROWS  (NBATCH * LSEQ)   // 8192
#define CHUNK  256
#define NCHK   (LSEQ / CHUNK)    // 8

// ---------------- WMMA helpers ----------------

__device__ __forceinline__ v8f wmma_f32_f16(v16h a, v16h b, v8f c) {
  return __builtin_amdgcn_wmma_f32_16x16x32_f16(
      /*neg_a=*/false, a, /*neg_b=*/false, b,
      /*c_mod=*/(short)0, c, /*reuse_a=*/false, /*reuse_b=*/false);
}

// 16x32 f16 fragment from row-major [rows, ld]; lane = row (m = lane&15),
// lanes 16-31 take K+8; VGPR v<4 -> K=2v,2v+1 ; v>=4 -> K=16+2(v-4),+1.
__device__ __forceinline__ v16h load_frag(const _Float16* __restrict__ base,
                                          int ld, int r0, int k0, int lane) {
  const int r  = r0 + (lane & 15);
  const int kh = (lane >> 4) << 3;
  const _Float16* p = base + (size_t)r * ld + (size_t)(k0 + kh);
  v16h f;
#pragma unroll
  for (int v = 0; v < 8; ++v) {
    const int kb = (v < 4) ? (v << 1) : (16 + ((v - 4) << 1));
    h2v pr = *(const h2v*)(p + kb);
    f[2 * v]     = pr[0];
    f[2 * v + 1] = pr[1];
  }
  return f;
}

// C/D tile: vgpr r -> row m0 + r + 8*(lane>=16), col n0 + (lane&15)
__device__ __forceinline__ void store_tile_f32(float* __restrict__ C, int ldC,
                                               int m0, int n0, const v8f& acc,
                                               int lane) {
  const int n  = n0 + (lane & 15);
  const int mb = m0 + ((lane >> 4) << 3);
#pragma unroll
  for (int r = 0; r < 8; ++r) C[(size_t)(mb + r) * ldC + n] = acc[r];
}

// ---------------- conversion kernels ----------------

__global__ void k_cvt_f16(const float* __restrict__ s, _Float16* __restrict__ d,
                          size_t n) {
  size_t i = (size_t)blockIdx.x * blockDim.x + threadIdx.x;
  if (i < n) d[i] = (_Float16)s[i];
}

// src [R, C] f32 row-major  ->  dst [C, R] f16 row-major (ld = R)
__global__ void k_transpose_cvt(const float* __restrict__ s,
                                _Float16* __restrict__ d, int R, int C) {
  size_t i = (size_t)blockIdx.x * blockDim.x + threadIdx.x;
  if (i < (size_t)R * C) {
    int r = (int)(i / C), c = (int)(i % C);
    d[(size_t)c * R + r] = (_Float16)s[i];
  }
}

// ---------------- GEMM kernels ----------------

// in_proj + gated silu: 2 m-tiles x 1 n-pair (gate,in) per wave.
__global__ __launch_bounds__(256) void k_gemm_silu(
    const _Float16* __restrict__ X,   // [MROWS, DMODEL]
    const _Float16* __restrict__ Wt,  // [2*DINNER, DMODEL]
    float* __restrict__ XS) {         // [MROWS, DINNER]
  const int lane = threadIdx.x & 31;
  const int wave = blockIdx.x * (blockDim.x >> 5) + (threadIdx.x >> 5);
  const int mt = MROWS >> 5, nt = DINNER >> 4;   // 256 x 128
  if (wave >= mt * nt) return;
  const int m0 = (wave % mt) << 5;
  const int n0 = (wave / mt) << 4;
  v8f cg0 = {}, cg1 = {}, ci0 = {}, ci1 = {};
  for (int k = 0; k < DMODEL; k += 32) {
    v16h a0 = load_frag(X,  DMODEL, m0,      k, lane);
    v16h a1 = load_frag(X,  DMODEL, m0 + 16, k, lane);
    v16h bg = load_frag(Wt, DMODEL, n0,          k, lane);
    v16h bi = load_frag(Wt, DMODEL, DINNER + n0, k, lane);
    cg0 = wmma_f32_f16(a0, bg, cg0);
    cg1 = wmma_f32_f16(a1, bg, cg1);
    ci0 = wmma_f32_f16(a0, bi, ci0);
    ci1 = wmma_f32_f16(a1, bi, ci1);
  }
  const int n = n0 + (lane & 15);
#pragma unroll
  for (int t = 0; t < 2; ++t) {
    const v8f& cg = t ? cg1 : cg0;
    const v8f& ci = t ? ci1 : ci0;
    const int mb = m0 + 16 * t + ((lane >> 4) << 3);
#pragma unroll
    for (int r = 0; r < 8; ++r) {
      float g = cg[r];
      float s = g / (1.f + __expf(-g));          // silu
      XS[(size_t)(mb + r) * DINNER + n] = s * ci[r];
    }
  }
}

// delta = softplus(x @ W_delta + b): 2x2 tiles per wave
__global__ __launch_bounds__(256) void k_gemm_delta(
    const _Float16* __restrict__ X, const _Float16* __restrict__ Wt,
    const float* __restrict__ bias, float* __restrict__ DLT) {
  const int lane = threadIdx.x & 31;
  const int wave = blockIdx.x * (blockDim.x >> 5) + (threadIdx.x >> 5);
  const int mt = MROWS >> 5, nt = DINNER >> 5;   // 256 x 64
  if (wave >= mt * nt) return;
  const int m0 = (wave % mt) << 5;
  const int n0 = (wave / mt) << 5;
  v8f c00 = {}, c01 = {}, c10 = {}, c11 = {};
  for (int k = 0; k < DMODEL; k += 32) {
    v16h a0 = load_frag(X,  DMODEL, m0,      k, lane);
    v16h a1 = load_frag(X,  DMODEL, m0 + 16, k, lane);
    v16h b0 = load_frag(Wt, DMODEL, n0,      k, lane);
    v16h b1 = load_frag(Wt, DMODEL, n0 + 16, k, lane);
    c00 = wmma_f32_f16(a0, b0, c00);
    c01 = wmma_f32_f16(a0, b1, c01);
    c10 = wmma_f32_f16(a1, b0, c10);
    c11 = wmma_f32_f16(a1, b1, c11);
  }
#pragma unroll
  for (int tm = 0; tm < 2; ++tm) {
#pragma unroll
    for (int tn = 0; tn < 2; ++tn) {
      const v8f& acc = tm ? (tn ? c11 : c10) : (tn ? c01 : c00);
      const int n  = n0 + 16 * tn + (lane & 15);
      const int mb = m0 + 16 * tm + ((lane >> 4) << 3);
      const float bv = bias[n];
#pragma unroll
      for (int r = 0; r < 8; ++r) {
        float v  = acc[r] + bv;
        float sp = fmaxf(v, 0.f) + log1pf(__expf(-fabsf(v)));
        DLT[(size_t)(mb + r) * DINNER + n] = sp;
      }
    }
  }
}

// plain GEMM, one 16x16 tile per wave (small-N cases: B/C projection)
__global__ __launch_bounds__(256) void k_gemm_plain(
    const _Float16* __restrict__ X, const _Float16* __restrict__ Wt,
    float* __restrict__ C, int M, int N, int K) {
  const int lane = threadIdx.x & 31;
  const int wave = blockIdx.x * (blockDim.x >> 5) + (threadIdx.x >> 5);
  const int mt = M >> 4, nt = N >> 4;
  if (wave >= mt * nt) return;
  const int m0 = (wave % mt) << 4;
  const int n0 = (wave / mt) << 4;
  v8f acc = {};
  for (int k = 0; k < K; k += 32) {
    v16h a = load_frag(X,  K, m0, k, lane);
    v16h b = load_frag(Wt, K, n0, k, lane);
    acc = wmma_f32_f16(a, b, acc);
  }
  store_tile_f32(C, N, m0, n0, acc, lane);
}

// plain GEMM, 2x2 tiles per wave (out_proj)
__global__ __launch_bounds__(256) void k_gemm_plain2x2(
    const _Float16* __restrict__ X, const _Float16* __restrict__ Wt,
    float* __restrict__ C, int M, int N, int K) {
  const int lane = threadIdx.x & 31;
  const int wave = blockIdx.x * (blockDim.x >> 5) + (threadIdx.x >> 5);
  const int mt = M >> 5, nt = N >> 5;
  if (wave >= mt * nt) return;
  const int m0 = (wave % mt) << 5;
  const int n0 = (wave / mt) << 5;
  v8f c00 = {}, c01 = {}, c10 = {}, c11 = {};
  for (int k = 0; k < K; k += 32) {
    v16h a0 = load_frag(X,  K, m0,      k, lane);
    v16h a1 = load_frag(X,  K, m0 + 16, k, lane);
    v16h b0 = load_frag(Wt, K, n0,      k, lane);
    v16h b1 = load_frag(Wt, K, n0 + 16, k, lane);
    c00 = wmma_f32_f16(a0, b0, c00);
    c01 = wmma_f32_f16(a0, b1, c01);
    c10 = wmma_f32_f16(a1, b0, c10);
    c11 = wmma_f32_f16(a1, b1, c11);
  }
  store_tile_f32(C, N, m0,      n0,      c00, lane);
  store_tile_f32(C, N, m0,      n0 + 16, c01, lane);
  store_tile_f32(C, N, m0 + 16, n0,      c10, lane);
  store_tile_f32(C, N, m0 + 16, n0 + 16, c11, lane);
}

// ---------------- chunked selective scan ----------------
// Phase A: per-chunk local scan (h0 = 0) with fused depthwise conv.
// Records P = prod(dA) and final local h per (b, chunk, d, n).
__global__ __launch_bounds__(256) void k_scan_local(
    const float* __restrict__ XS, const float* __restrict__ DLT,
    const float* __restrict__ BC, const float* __restrict__ conv_w,
    const float* __restrict__ conv_b, const float* __restrict__ A_log,
    const float* __restrict__ Dvec, float* __restrict__ YLOC,
    float* __restrict__ Pprod, float* __restrict__ Hfin) {
  const int blocksPerD = DINNER / 256;           // 8
  int t = blockIdx.x;
  const int dblk = t % blocksPerD; t /= blocksPerD;
  const int c = t % NCHK;          t /= NCHK;
  const int b = t;
  const int d  = dblk * 256 + threadIdx.x;
  const int l0 = c * CHUNK;

  const float w0 = conv_w[d * 4 + 0], w1 = conv_w[d * 4 + 1];
  const float w2 = conv_w[d * 4 + 2], w3 = conv_w[d * 4 + 3];
  const float cb = conv_b[d];
  const float Dd = Dvec[d];

  float a[NSTATE], h[NSTATE], p[NSTATE];
#pragma unroll
  for (int n = 0; n < NSTATE; ++n) {
    a[n] = -__expf(A_log[(size_t)d * NSTATE + n]);
    h[n] = 0.f;
    p[n] = 1.f;
  }

  const size_t rowBase = (size_t)b * LSEQ;
  float x0 = 0.f, x1 = 0.f, x2 = 0.f;
  if (c != 0) {                                  // seed conv window across chunk edge
    x0 = XS[(rowBase + l0 - 3) * DINNER + d];
    x1 = XS[(rowBase + l0 - 2) * DINNER + d];
    x2 = XS[(rowBase + l0 - 1) * DINNER + d];
  }

  __shared__ float sBC[2][32];
  if (threadIdx.x < 32) sBC[0][threadIdx.x] = BC[(rowBase + l0) * 32 + threadIdx.x];
  __syncthreads();

  size_t idx = (rowBase + l0) * DINNER + d;
  float dl = DLT[idx];
  float xs = XS[idx];

  for (int i = 0; i < CHUNK; ++i) {
    const int cur = i & 1;
    if (i + 1 < CHUNK && threadIdx.x < 32)
      sBC[cur ^ 1][threadIdx.x] = BC[(rowBase + l0 + i + 1) * 32 + threadIdx.x];
    float dln = 0.f, xsn = 0.f;
    if (i + 1 < CHUNK) { dln = DLT[idx + DINNER]; xsn = XS[idx + DINNER]; }

    const float xv = w0 * x0 + w1 * x1 + w2 * x2 + w3 * xs + cb;
    x0 = x1; x1 = x2; x2 = xs;

    const float dx = dl * xv;
    float y = Dd * xv;
#pragma unroll
    for (int n = 0; n < NSTATE; ++n) {
      const float dA = __expf(dl * a[n]);
      p[n] *= dA;
      h[n] = dA * h[n] + dx * sBC[cur][n];
      y   += sBC[cur][16 + n] * h[n];
    }
    YLOC[idx] = y;

    __syncthreads();
    dl = dln; xs = xsn; idx += DINNER;
  }

  const size_t sbase = (((size_t)b * NCHK + c) * DINNER + d) * NSTATE;
#pragma unroll
  for (int n = 0; n < NSTATE; ++n) { Pprod[sbase + n] = p[n]; Hfin[sbase + n] = h[n]; }
}

// Phase B: sequentially combine chunk boundary states per channel.
// Hin(0)=0 ; Hin(c) = Hfin(c-1) + P(c-1) * Hin(c-1)
__global__ __launch_bounds__(256) void k_scan_combine(
    const float* __restrict__ Pprod, const float* __restrict__ Hfin,
    float* __restrict__ Hin) {
  const int t = blockIdx.x * blockDim.x + threadIdx.x;   // 0..NBATCH*DINNER-1
  const int b = t / DINNER, d = t % DINNER;
  float hc[NSTATE];
#pragma unroll
  for (int n = 0; n < NSTATE; ++n) hc[n] = 0.f;
  for (int c = 0; c < NCHK; ++c) {
    const size_t base = (((size_t)b * NCHK + c) * DINNER + d) * NSTATE;
#pragma unroll
    for (int n = 0; n < NSTATE; ++n) {
      Hin[base + n] = hc[n];
      hc[n] = Hfin[base + n] + Pprod[base + n] * hc[n];
    }
  }
}

// Phase C: add correction  y_t += C_t . (Pcum_t * Hin)  and emit f16.
__global__ __launch_bounds__(256) void k_scan_fix(
    const float* __restrict__ DLT, const float* __restrict__ BC,
    const float* __restrict__ A_log, const float* __restrict__ YLOC,
    const float* __restrict__ Hin, _Float16* __restrict__ Y16) {
  const int blocksPerD = DINNER / 256;
  int t = blockIdx.x;
  const int dblk = t % blocksPerD; t /= blocksPerD;
  const int c = t % NCHK;          t /= NCHK;
  const int b = t;
  const int d  = dblk * 256 + threadIdx.x;
  const int l0 = c * CHUNK;
  const size_t rowBase = (size_t)b * LSEQ;
  size_t idx = (rowBase + l0) * DINNER + d;

  if (c == 0) {                                  // Hin = 0: straight convert
    for (int i = 0; i < CHUNK; ++i) { Y16[idx] = (_Float16)YLOC[idx]; idx += DINNER; }
    return;
  }

  float a[NSTATE], q[NSTATE];
  const size_t sbase = (((size_t)b * NCHK + c) * DINNER + d) * NSTATE;
#pragma unroll
  for (int n = 0; n < NSTATE; ++n) {
    a[n] = -__expf(A_log[(size_t)d * NSTATE + n]);
    q[n] = Hin[sbase + n];                       // q = Pcum * Hin (running)
  }

  __shared__ float sC[2][16];
  if (threadIdx.x < 16) sC[0][threadIdx.x] = BC[(rowBase + l0) * 32 + 16 + threadIdx.x];
  __syncthreads();

  float dl = DLT[idx];
  for (int i = 0; i < CHUNK; ++i) {
    const int cur = i & 1;
    if (i + 1 < CHUNK && threadIdx.x < 16)
      sC[cur ^ 1][threadIdx.x] = BC[(rowBase + l0 + i + 1) * 32 + 16 + threadIdx.x];
    const float dln = (i + 1 < CHUNK) ? DLT[idx + DINNER] : 0.f;

    float y = YLOC[idx];
#pragma unroll
    for (int n = 0; n < NSTATE; ++n) {
      q[n] *= __expf(dl * a[n]);
      y    += sC[cur][n] * q[n];
    }
    Y16[idx] = (_Float16)y;

    __syncthreads();
    dl = dln; idx += DINNER;
  }
}

// ---------------- launcher ----------------

extern "C" void kernel_launch(void* const* d_in, const int* in_sizes, int n_in,
                              void* d_out, int out_size, void* d_ws, size_t ws_size,
                              hipStream_t stream) {
  const float* x       = (const float*)d_in[0];
  const float* W_in    = (const float*)d_in[1];
  const float* conv_w  = (const float*)d_in[2];
  const float* conv_b  = (const float*)d_in[3];
  const float* W_delta = (const float*)d_in[4];
  const float* b_delta = (const float*)d_in[5];
  const float* W_B     = (const float*)d_in[6];
  const float* W_C     = (const float*)d_in[7];
  const float* W_out   = (const float*)d_in[8];
  const float* A_log   = (const float*)d_in[9];
  const float* Dvec    = (const float*)d_in[10];
  float* out = (float*)d_out;

  char* ws = (char*)d_ws;
  size_t off = 0;
  auto alloc = [&](size_t bytes) -> char* {
    char* p = ws + off;
    off = (off + bytes + 255) & ~(size_t)255;
    return p;
  };
  _Float16* x16    = (_Float16*)alloc((size_t)MROWS * DMODEL * 2);
  _Float16* Win_t  = (_Float16*)alloc((size_t)2 * DINNER * DMODEL * 2);
  _Float16* Wd_t   = (_Float16*)alloc((size_t)DINNER * DMODEL * 2);
  _Float16* WBC_t  = (_Float16*)alloc((size_t)32 * DMODEL * 2);
  _Float16* Wout_t = (_Float16*)alloc((size_t)DMODEL * DINNER * 2);
  float*    XS     = (float*)   alloc((size_t)MROWS * DINNER * 4);
  float*    DLT    = (float*)   alloc((size_t)MROWS * DINNER * 4);
  float*    BCbuf  = (float*)   alloc((size_t)MROWS * 32 * 4);
  float*    YLOC   = (float*)   alloc((size_t)MROWS * DINNER * 4);
  _Float16* Y16    = (_Float16*)alloc((size_t)MROWS * DINNER * 2);
  const size_t sumN = (size_t)NBATCH * NCHK * DINNER * NSTATE;
  float*    Pprod  = (float*)   alloc(sumN * 4);
  float*    Hfin   = (float*)   alloc(sumN * 4);
  float*    HinB   = (float*)   alloc(sumN * 4);
  (void)ws_size; (void)n_in; (void)in_sizes; (void)out_size;

  const int T = 256;
  auto blk1 = [&](size_t n) { return (unsigned)((n + T - 1) / T); };

  // operand conversion / weight transposition (one-time, f32 -> f16)
  k_cvt_f16<<<blk1((size_t)MROWS * DMODEL), T, 0, stream>>>(x, x16, (size_t)MROWS * DMODEL);
  k_transpose_cvt<<<blk1((size_t)DMODEL * 2 * DINNER), T, 0, stream>>>(W_in, Win_t, DMODEL, 2 * DINNER);
  k_transpose_cvt<<<blk1((size_t)DMODEL * DINNER), T, 0, stream>>>(W_delta, Wd_t, DMODEL, DINNER);
  k_transpose_cvt<<<blk1((size_t)DMODEL * 16), T, 0, stream>>>(W_B, WBC_t, DMODEL, 16);
  k_transpose_cvt<<<blk1((size_t)DMODEL * 16), T, 0, stream>>>(W_C, WBC_t + (size_t)16 * DMODEL, DMODEL, 16);
  k_transpose_cvt<<<blk1((size_t)DINNER * DMODEL), T, 0, stream>>>(W_out, Wout_t, DINNER, DMODEL);

  // in_proj + silu gate (WMMA, 2m x (gate,in) blocking)
  k_gemm_silu<<<(MROWS / 32) * (DINNER / 16) / 8, 256, 0, stream>>>(x16, Win_t, XS);
  // delta projection (WMMA, 2x2 blocking)
  k_gemm_delta<<<(MROWS / 32) * (DINNER / 32) / 8, 256, 0, stream>>>(x16, Wd_t, b_delta, DLT);
  // B/C projections (WMMA)
  k_gemm_plain<<<(MROWS / 16) * (32 / 16) / 8, 256, 0, stream>>>(x16, WBC_t, BCbuf, MROWS, 32, DMODEL);

  // chunked scan: local -> combine -> fix
  const unsigned scanBlocks = NBATCH * NCHK * (DINNER / 256);   // 256
  k_scan_local<<<scanBlocks, 256, 0, stream>>>(XS, DLT, BCbuf, conv_w, conv_b,
                                               A_log, Dvec, YLOC, Pprod, Hfin);
  k_scan_combine<<<(NBATCH * DINNER) / 256, 256, 0, stream>>>(Pprod, Hfin, HinB);
  k_scan_fix<<<scanBlocks, 256, 0, stream>>>(DLT, BCbuf, A_log, YLOC, HinB, Y16);

  // out_proj (WMMA, 2x2 blocking)
  k_gemm_plain2x2<<<(MROWS / 32) * (DMODEL / 32) / 8, 256, 0, stream>>>(
      Y16, Wout_t, out, MROWS, DMODEL, DINNER);
}
